// Decoder_77610059039330
// MI455X (gfx1250) — matrix operand order repaired and checked by reading, not compiled, not run on hardware
//
#include <hip/hip_runtime.h>
#include <math.h>

// Problem constants (match reference)
#define B_   4
#define LD_  1024
#define LE_  1024
#define D_   256
#define H_   8
#define DK_  32
#define DFF_ 1024
#define NL_  6
#define M_   (B_ * LD_)   // 4096 rows (B*L)

typedef __attribute__((ext_vector_type(16))) __bf16 v16bf;
typedef __attribute__((ext_vector_type(8)))  float  v8f;

union BFV {
  v16bf v;
  unsigned short u[16];
  uint4 q[2];
};

__device__ __forceinline__ unsigned short f2bf(float f) {
  unsigned u = __float_as_uint(f);
  u += 0x7FFFu + ((u >> 16) & 1u);   // round-to-nearest-even
  return (unsigned short)(u >> 16);
}

// ---------------------------------------------------------------------------
// fp32 -> bf16 elementwise convert (row-major, for activations)
// ---------------------------------------------------------------------------
__global__ void __launch_bounds__(256) cvt_bf16_kernel(const float* __restrict__ in,
                                                       unsigned short* __restrict__ out,
                                                       int n) {
  int i = blockIdx.x * 256 + threadIdx.x;
  if (i < n) out[i] = f2bf(in[i]);
}

// ---------------------------------------------------------------------------
// fp32 [NL,K,N] -> bf16 transposed [NL,N,K] (weights: B operand wants W^T rows)
// ---------------------------------------------------------------------------
__global__ void __launch_bounds__(256) cvt_t_bf16_kernel(const float* __restrict__ in,
                                                         unsigned short* __restrict__ out,
                                                         int K, int N, int total) {
  int idx = blockIdx.x * 256 + threadIdx.x;
  if (idx >= total) return;
  int per = K * N;
  int layer = idx / per;
  int rem = idx - layer * per;
  int k = rem / N;
  int n = rem - k * N;
  out[(size_t)layer * per + (size_t)n * K + k] = f2bf(in[idx]);
}

// ---------------------------------------------------------------------------
// x = dec_inputs @ W_tgt + b_tgt   (K=3, trivial)
// ---------------------------------------------------------------------------
__global__ void __launch_bounds__(256) embed_kernel(const float* __restrict__ dec_in,
                                                    const float* __restrict__ Wt,
                                                    const float* __restrict__ bt,
                                                    float* __restrict__ x) {
  int idx = blockIdx.x * 256 + threadIdx.x;       // over M_*D_
  int d = idx & (D_ - 1);
  int m = idx >> 8;
  float v = bt[d];
#pragma unroll
  for (int c = 0; c < 3; ++c) v += dec_in[m * 3 + c] * Wt[c * D_ + d];
  x[idx] = v;
}

// ---------------------------------------------------------------------------
// x += fixed positional embedding; also emit bf16 copy
// ---------------------------------------------------------------------------
__global__ void __launch_bounds__(256) add_pe_kernel(float* __restrict__ x,
                                                     unsigned short* __restrict__ xbf) {
  int idx = blockIdx.x * 256 + threadIdx.x;       // over M_*D_
  int d = idx & (D_ - 1);
  int l = (idx >> 8) & (LD_ - 1);
  int i = d & 127;
  float invf = __expf(-9.210340371976184f * (float)(2 * i) / 256.0f); // 10000^{-2i/256}
  float ang = (float)l * invf;
  float pe = (d < 128) ? sinf(ang) : cosf(ang);
  float v = x[idx] + pe;
  x[idx] = v;
  xbf[idx] = f2bf(v);
}

// ---------------------------------------------------------------------------
// LayerNorm over D=256 per row; writes f32 and bf16 copies
// ---------------------------------------------------------------------------
__global__ void __launch_bounds__(256) ln_kernel(const float* __restrict__ in,
                                                 const float* __restrict__ g,
                                                 const float* __restrict__ b,
                                                 float* __restrict__ outf,
                                                 unsigned short* __restrict__ outbf) {
  __shared__ float red[256];
  int row = blockIdx.x;
  int t = threadIdx.x;
  float v = in[(size_t)row * D_ + t];
  red[t] = v;
  __syncthreads();
  for (int s = 128; s > 0; s >>= 1) { if (t < s) red[t] += red[t + s]; __syncthreads(); }
  float mean = red[0] * (1.0f / D_);
  __syncthreads();
  float c = v - mean;
  red[t] = c * c;
  __syncthreads();
  for (int s = 128; s > 0; s >>= 1) { if (t < s) red[t] += red[t + s]; __syncthreads(); }
  float var = red[0] * (1.0f / D_);
  float y = c * rsqrtf(var + 1e-5f) * g[t] + b[t];
  outf[(size_t)row * D_ + t] = y;
  outbf[(size_t)row * D_ + t] = f2bf(y);
}

// ---------------------------------------------------------------------------
// GEMM epilogue helper
// outbfT writes [B, N, 1024] "transposed" bf16 (for attention V operand)
// ---------------------------------------------------------------------------
__device__ __forceinline__ void emit_out(float v, int m, int n, int N,
                                         const float* resid, int relu,
                                         float* outf, unsigned short* outbf,
                                         unsigned short* outbfT) {
  if (resid) v += resid[(size_t)m * N + n];
  if (relu) v = fmaxf(v, 0.f);
  if (outf)  outf[(size_t)m * N + n] = v;
  if (outbf) outbf[(size_t)m * N + n] = f2bf(v);
  if (outbfT) outbfT[((size_t)((m >> 10) * N + n) << 10) + (m & 1023)] = f2bf(v);
}

// ---------------------------------------------------------------------------
// GEMM: C[M,N] = A[M,K](bf16) @ W[K,N] + bias (+resid) (relu?)
// W is pre-transposed: Wt[N,K] so B operands are contiguous.
// 256 threads = 8 waves; block tile 128x64 (waves 4M x 2N); K stepped by 64
// through DOUBLE-BUFFERED LDS: one barrier per 8 WMMAs.
// ---------------------------------------------------------------------------
__global__ void __launch_bounds__(256) gemm_bf16_wmma(
    const unsigned short* __restrict__ A,    // [M,K] bf16 row-major
    const unsigned short* __restrict__ Wt,   // [N,K] bf16 row-major (W^T)
    const float* __restrict__ bias,          // [N]
    const float* __restrict__ resid,         // [M,N] f32 or null
    float* __restrict__ outf,                // [M,N] f32 or null
    unsigned short* __restrict__ outbf,      // [M,N] bf16 or null
    unsigned short* __restrict__ outbfT,     // [B,N,1024] bf16 or null
    int M, int N, int K, int relu) {
  __shared__ unsigned short sA[2][128 * 64]; // 2 x 16 KB
  __shared__ unsigned short sB[2][64 * 64];  // 2 x  8 KB  (48 KB total)
  const int tid = threadIdx.x;
  const int lane = tid & 31;
  const int wave = tid >> 5;
  const int mT = wave & 3;                   // 4 M-subtiles of 32 rows
  const int nT = wave >> 2;                  // 2 N-subtiles of 32 cols
  const int lh = lane & 15, lg = lane >> 4;
  const int blockM = blockIdx.y * 128;
  const int blockN = blockIdx.x * 64;

  v8f c00 = {0.f,0.f,0.f,0.f,0.f,0.f,0.f,0.f};
  v8f c01 = c00, c10 = c00, c11 = c00;

  // cooperative staging: A 128x64 (32 ushort/thread), Wt 64x64 (16 ushort/thread)
  const int arow = tid >> 1;
  const int acol = (tid & 1) * 32;
  const int brow = tid >> 2;
  const int bcol = (tid & 3) * 16;
  const unsigned short* Asrc = A + (size_t)(blockM + arow) * K + acol;
  const unsigned short* Bsrc = Wt + (size_t)(blockN + brow) * K + bcol;

  // preload stage 0
  {
    unsigned short* dA = sA[0] + arow * 64 + acol;
    unsigned short* dB = sB[0] + brow * 64 + bcol;
    *(uint4*)(dA)      = *(const uint4*)(Asrc);
    *(uint4*)(dA + 8)  = *(const uint4*)(Asrc + 8);
    *(uint4*)(dA + 16) = *(const uint4*)(Asrc + 16);
    *(uint4*)(dA + 24) = *(const uint4*)(Asrc + 24);
    *(uint4*)(dB)      = *(const uint4*)(Bsrc);
    *(uint4*)(dB + 8)  = *(const uint4*)(Bsrc + 8);
  }
  __syncthreads();

  int cur = 0;
  for (int k0 = 0; k0 < K; k0 += 64) {
    // stage next K-slab into the other buffer while computing this one
    if (k0 + 64 < K) {
      const unsigned short* as = Asrc + k0 + 64;
      const unsigned short* bs = Bsrc + k0 + 64;
      unsigned short* dA = sA[cur ^ 1] + arow * 64 + acol;
      unsigned short* dB = sB[cur ^ 1] + brow * 64 + bcol;
      *(uint4*)(dA)      = *(const uint4*)(as);
      *(uint4*)(dA + 8)  = *(const uint4*)(as + 8);
      *(uint4*)(dA + 16) = *(const uint4*)(as + 16);
      *(uint4*)(dA + 24) = *(const uint4*)(as + 24);
      *(uint4*)(dB)      = *(const uint4*)(bs);
      *(uint4*)(dB + 8)  = *(const uint4*)(bs + 8);
      if (k0 + 128 < K) {           // prefetch the slab after next
        __builtin_prefetch(as + 64, 0, 3);
        __builtin_prefetch(bs + 64, 0, 3);
      }
    }
#pragma unroll
    for (int kk = 0; kk < 64; kk += 32) {
      BFV a0, a1, b0, b1;
      // A operands: 32B contiguous per lane (k = 16*lg + s), matching B
      const unsigned short* pa0 = sA[cur] + (mT * 32 + lh) * 64 + kk;
      const unsigned short* pa1 = pa0 + 16 * 64;
      const int koff = lg * 16;
      a0.q[0] = *(const uint4*)(pa0 + koff);
      a0.q[1] = *(const uint4*)(pa0 + 8 + koff);
      a1.q[0] = *(const uint4*)(pa1 + koff);
      a1.q[1] = *(const uint4*)(pa1 + 8 + koff);
      // B operands: Wt rows contiguous in K -> 32B contiguous per lane
      const unsigned short* pb0 = sB[cur] + (nT * 32 + lh) * 64 + kk + lg * 16;
      const unsigned short* pb1 = pb0 + 16 * 64;
      b0.q[0] = *(const uint4*)(pb0);
      b0.q[1] = *(const uint4*)(pb0 + 8);
      b1.q[0] = *(const uint4*)(pb1);
      b1.q[1] = *(const uint4*)(pb1 + 8);

      c00 = __builtin_amdgcn_wmma_f32_16x16x32_bf16(false, a0.v, false, b0.v,
                                                    (short)0, c00, false, false);
      c01 = __builtin_amdgcn_wmma_f32_16x16x32_bf16(false, a0.v, false, b1.v,
                                                    (short)0, c01, false, false);
      c10 = __builtin_amdgcn_wmma_f32_16x16x32_bf16(false, a1.v, false, b0.v,
                                                    (short)0, c10, false, false);
      c11 = __builtin_amdgcn_wmma_f32_16x16x32_bf16(false, a1.v, false, b1.v,
                                                    (short)0, c11, false, false);
    }
    __syncthreads();   // single barrier per K-step (ping-pong buffers)
    cur ^= 1;
  }

  // C/D layout: VGPR r -> M = r (+8 for lanes 16-31); N = lane%16
  const int n0 = blockN + nT * 32 + lh;
  const int n1 = n0 + 16;
  const float bv0 = bias ? bias[n0] : 0.f;
  const float bv1 = bias ? bias[n1] : 0.f;
#pragma unroll
  for (int r = 0; r < 8; ++r) {
    const int m0 = blockM + mT * 32 + r + lg * 8;
    const int m1 = m0 + 16;
    emit_out(c00[r] + bv0, m0, n0, N, resid, relu, outf, outbf, outbfT);
    emit_out(c01[r] + bv1, m0, n1, N, resid, relu, outf, outbf, outbfT);
    emit_out(c10[r] + bv0, m1, n0, N, resid, relu, outf, outbf, outbfT);
    emit_out(c11[r] + bv1, m1, n1, N, resid, relu, outf, outbf, outbfT);
  }
}

// ---------------------------------------------------------------------------
// Flash attention: wave handles 16 query rows of one (b,h). DK=DV=32.
// Per 32-key chunk: S = Q@K^T (2 WMMA), mask + online softmax, ctx += P@V
// (2 WMMA). V comes pre-transposed (Vt[(b*256+d)][l]) so its B operand is
// contiguous global 16B loads; only P goes through LDS (cross-lane relayout).
// ---------------------------------------------------------------------------
__global__ void __launch_bounds__(128) attn_wmma(
    const unsigned short* __restrict__ Qbf,   // [B*LQ, 256]
    const unsigned short* __restrict__ Kbf,   // [B*LK, 256]
    const unsigned short* __restrict__ Vt,    // [B*256, LK(=1024)]
    const int* __restrict__ kmask,            // [B, LK] pad mask over keys
    unsigned short* __restrict__ ctxbf,       // [B*LQ, 256]
    int LQ, int LK, int causal, float scale) {
  __shared__ unsigned short sP[4][16 * 32];
  const int lane = threadIdx.x & 31;
  const int wave = threadIdx.x >> 5;
  const int lh = lane & 15, lg = lane >> 4;
  const int q0 = (blockIdx.x * 4 + wave) * 16;
  const int h = blockIdx.y, b = blockIdx.z;

  unsigned short* myP = sP[wave];

  // Q as WMMA A operand (16 queries x 32 dk)
  BFV qa;
  {
    const unsigned short* p = Qbf + (size_t)(b * LQ + q0 + lh) * 256 + h * 32;
    qa.q[0] = *(const uint4*)(p + lg * 16);
    qa.q[1] = *(const uint4*)(p + lg * 16 + 8);
  }
  // Vt row bases for the two dv halves (dv = lh and lh+16)
  const unsigned short* vrow0 = Vt + ((size_t)(b * 256 + h * 32 + lh) << 10);
  const unsigned short* vrow1 = vrow0 + ((size_t)16 << 10);

  v8f acc0 = {0.f,0.f,0.f,0.f,0.f,0.f,0.f,0.f};
  v8f acc1 = acc0;
  float mrow[8], lrow[8];
#pragma unroll
  for (int r = 0; r < 8; ++r) { mrow[r] = -1e30f; lrow[r] = 0.f; }

  for (int kc = 0; kc < LK; kc += 32) {
    if (causal && kc > q0 + 15) break;   // wave-uniform early exit

    // K^T B operands: B[k][n] = K[key kc+n][dk k] -> contiguous per lane
    BFV bk0, bk1;
    {
      const unsigned short* p = Kbf + (size_t)(b * LK + kc + lh) * 256 + h * 32 + lg * 16;
      bk0.q[0] = *(const uint4*)(p);
      bk0.q[1] = *(const uint4*)(p + 8);
      p += (size_t)16 * 256;
      bk1.q[0] = *(const uint4*)(p);
      bk1.q[1] = *(const uint4*)(p + 8);
    }

    v8f z = {0.f,0.f,0.f,0.f,0.f,0.f,0.f,0.f};
    v8f s0 = __builtin_amdgcn_wmma_f32_16x16x32_bf16(false, qa.v, false, bk0.v,
                                                     (short)0, z, false, false);
    v8f s1 = __builtin_amdgcn_wmma_f32_16x16x32_bf16(false, qa.v, false, bk1.v,
                                                     (short)0, z, false, false);

    const int n0 = kc + lh, n1 = kc + 16 + lh;
    const int pm0 = (kmask[(size_t)b * LK + n0] != 0);
    const int pm1 = (kmask[(size_t)b * LK + n1] != 0);

#pragma unroll
    for (int r = 0; r < 8; ++r) {
      const int qrow = q0 + r + lg * 8;
      float v0 = s0[r] * scale;
      float v1 = s1[r] * scale;
      if (pm0 || (causal && n0 > qrow)) v0 = -1e9f;
      if (pm1 || (causal && n1 > qrow)) v1 = -1e9f;
      float t = fmaxf(v0, v1);
#pragma unroll
      for (int off = 8; off >= 1; off >>= 1) t = fmaxf(t, __shfl_xor(t, off, 16));
      const float mnew = fmaxf(mrow[r], t);
      const float p0 = __expf(v0 - mnew);
      const float p1 = __expf(v1 - mnew);
      float s = p0 + p1;
#pragma unroll
      for (int off = 8; off >= 1; off >>= 1) s += __shfl_xor(s, off, 16);
      const float fac = __expf(mrow[r] - mnew);
      lrow[r] = lrow[r] * fac + s;
      mrow[r] = mnew;
      acc0[r] *= fac;
      acc1[r] *= fac;
      const int prow = r + lg * 8;
      myP[prow * 32 + lh]      = f2bf(p0);
      myP[prow * 32 + 16 + lh] = f2bf(p1);
    }

    // P (16x32) regathered in A layout: 32B contiguous per lane (k = 16*lg + s)
    BFV pa, bv0, bv1;
    pa.q[0] = *(const uint4*)(myP + lh * 32 + lg * 16);
    pa.q[1] = *(const uint4*)(myP + lh * 32 + lg * 16 + 8);
    // V B operands straight from transposed global: contiguous along keys
    bv0.q[0] = *(const uint4*)(vrow0 + kc + lg * 16);
    bv0.q[1] = *(const uint4*)(vrow0 + kc + lg * 16 + 8);
    bv1.q[0] = *(const uint4*)(vrow1 + kc + lg * 16);
    bv1.q[1] = *(const uint4*)(vrow1 + kc + lg * 16 + 8);

    acc0 = __builtin_amdgcn_wmma_f32_16x16x32_bf16(false, pa.v, false, bv0.v,
                                                   (short)0, acc0, false, false);
    acc1 = __builtin_amdgcn_wmma_f32_16x16x32_bf16(false, pa.v, false, bv1.v,
                                                   (short)0, acc1, false, false);
  }

#pragma unroll
  for (int r = 0; r < 8; ++r) {
    const float inv = 1.0f / lrow[r];
    const size_t row = (size_t)(b * LQ + q0 + r + lg * 8) * 256 + h * 32;
    ctxbf[row + lh]      = f2bf(acc0[r] * inv);
    ctxbf[row + 16 + lh] = f2bf(acc1[r] * inv);
  }
}

// ---------------------------------------------------------------------------
// Host orchestration
// ---------------------------------------------------------------------------
extern "C" void kernel_launch(void* const* d_in, const int* in_sizes, int n_in,
                              void* d_out, int out_size, void* d_ws, size_t ws_size,
                              hipStream_t stream) {
  (void)in_sizes; (void)n_in; (void)out_size; (void)ws_size;

  const float* enc_outputs = (const float*)d_in[0];
  const float* dec_inputs  = (const float*)d_in[1];
  const int*   enc_masks   = (const int*)d_in[2];
  const int*   dec_masks   = (const int*)d_in[3];
  const float* W_tgt   = (const float*)d_in[4];
  const float* b_tgt   = (const float*)d_in[5];
  const float* sa_Wq   = (const float*)d_in[6];
  const float* sa_bq   = (const float*)d_in[7];
  const float* sa_Wk   = (const float*)d_in[8];
  const float* sa_bk   = (const float*)d_in[9];
  const float* sa_Wv   = (const float*)d_in[10];
  const float* sa_bv   = (const float*)d_in[11];
  const float* sa_Wo   = (const float*)d_in[12];
  const float* sa_bo   = (const float*)d_in[13];
  const float* sa_ln_g = (const float*)d_in[14];
  const float* sa_ln_b = (const float*)d_in[15];
  const float* ca_Wq   = (const float*)d_in[16];
  const float* ca_bq   = (const float*)d_in[17];
  const float* ca_Wk   = (const float*)d_in[18];
  const float* ca_bk   = (const float*)d_in[19];
  const float* ca_Wv   = (const float*)d_in[20];
  const float* ca_bv   = (const float*)d_in[21];
  const float* ca_Wo   = (const float*)d_in[22];
  const float* ca_bo   = (const float*)d_in[23];
  const float* ca_ln_g = (const float*)d_in[24];
  const float* ca_ln_b = (const float*)d_in[25];
  const float* ff_W1   = (const float*)d_in[26];
  const float* ff_b1   = (const float*)d_in[27];
  const float* ff_W2   = (const float*)d_in[28];
  const float* ff_b2   = (const float*)d_in[29];
  const float* ff_ln_g = (const float*)d_in[30];
  const float* ff_ln_b = (const float*)d_in[31];

  // Workspace carving (~40 MB total)
  char* p = (char*)d_ws;
  auto carve = [&](size_t bytes) -> char* {
    char* r = p;
    p += (bytes + 255) & ~(size_t)255;
    return r;
  };
  float* x   = (float*)carve((size_t)M_ * D_ * 4);
  float* tmp = (float*)carve((size_t)M_ * D_ * 4);
  unsigned short* xbf   = (unsigned short*)carve((size_t)M_ * D_ * 2);
  unsigned short* qbf   = (unsigned short*)carve((size_t)M_ * D_ * 2);
  unsigned short* kbf   = (unsigned short*)carve((size_t)M_ * D_ * 2);
  unsigned short* vtbf  = (unsigned short*)carve((size_t)M_ * D_ * 2); // transposed
  unsigned short* ctxbf = (unsigned short*)carve((size_t)M_ * D_ * 2);
  unsigned short* hbf   = (unsigned short*)carve((size_t)M_ * DFF_ * 2);
  unsigned short* encbf = (unsigned short*)carve((size_t)M_ * D_ * 2);
  unsigned short* wsaq = (unsigned short*)carve((size_t)NL_ * D_ * D_ * 2);
  unsigned short* wsak = (unsigned short*)carve((size_t)NL_ * D_ * D_ * 2);
  unsigned short* wsav = (unsigned short*)carve((size_t)NL_ * D_ * D_ * 2);
  unsigned short* wsao = (unsigned short*)carve((size_t)NL_ * D_ * D_ * 2);
  unsigned short* wcaq = (unsigned short*)carve((size_t)NL_ * D_ * D_ * 2);
  unsigned short* wcak = (unsigned short*)carve((size_t)NL_ * D_ * D_ * 2);
  unsigned short* wcav = (unsigned short*)carve((size_t)NL_ * D_ * D_ * 2);
  unsigned short* wcao = (unsigned short*)carve((size_t)NL_ * D_ * D_ * 2);
  unsigned short* wff1 = (unsigned short*)carve((size_t)NL_ * D_ * DFF_ * 2);
  unsigned short* wff2 = (unsigned short*)carve((size_t)NL_ * DFF_ * D_ * 2);

  // Weights converted AND transposed to [N,K] bf16 once per launch.
  auto cvtT = [&](const float* src, unsigned short* dst, int K, int N) {
    int total = NL_ * K * N;
    cvt_t_bf16_kernel<<<(total + 255) / 256, 256, 0, stream>>>(src, dst, K, N, total);
  };
  cvt_bf16_kernel<<<(M_ * D_ + 255) / 256, 256, 0, stream>>>(enc_outputs, encbf, M_ * D_);
  cvtT(sa_Wq, wsaq, D_, D_);
  cvtT(sa_Wk, wsak, D_, D_);
  cvtT(sa_Wv, wsav, D_, D_);
  cvtT(sa_Wo, wsao, D_, D_);
  cvtT(ca_Wq, wcaq, D_, D_);
  cvtT(ca_Wk, wcak, D_, D_);
  cvtT(ca_Wv, wcav, D_, D_);
  cvtT(ca_Wo, wcao, D_, D_);
  cvtT(ff_W1, wff1, D_, DFF_);
  cvtT(ff_W2, wff2, DFF_, D_);

  embed_kernel<<<(M_ * D_) / 256, 256, 0, stream>>>(dec_inputs, W_tgt, b_tgt, x);

  const float scale = 0.17677669529663687f;  // 1/sqrt(32)
  dim3 gO(D_ / 64, M_ / 128);                // (4, 32)   N=256 GEMMs
  dim3 gF(DFF_ / 64, M_ / 128);              // (16, 32)  N=1024 GEMM
  dim3 gA(LD_ / 64, H_, B_);                 // (16, 8, 4) attention

  for (int i = 0; i < NL_; ++i) {
    const size_t wo  = (size_t)i * D_ * D_;
    const size_t bo  = (size_t)i * D_;
    const size_t w1o = (size_t)i * D_ * DFF_;
    const size_t b1o = (size_t)i * DFF_;

    add_pe_kernel<<<(M_ * D_) / 256, 256, 0, stream>>>(x, xbf);

    // ---- self attention ----
    gemm_bf16_wmma<<<gO, 256, 0, stream>>>(xbf, wsaq + wo, sa_bq + bo, nullptr,
                                           nullptr, qbf, nullptr, M_, D_, D_, 0);
    gemm_bf16_wmma<<<gO, 256, 0, stream>>>(xbf, wsak + wo, sa_bk + bo, nullptr,
                                           nullptr, kbf, nullptr, M_, D_, D_, 0);
    gemm_bf16_wmma<<<gO, 256, 0, stream>>>(xbf, wsav + wo, sa_bv + bo, nullptr,
                                           nullptr, nullptr, vtbf, M_, D_, D_, 0);
    attn_wmma<<<gA, 128, 0, stream>>>(qbf, kbf, vtbf, dec_masks, ctxbf,
                                      LD_, LD_, 1, scale);
    gemm_bf16_wmma<<<gO, 256, 0, stream>>>(ctxbf, wsao + wo, sa_bo + bo, x,
                                           tmp, nullptr, nullptr, M_, D_, D_, 0);
    ln_kernel<<<M_, 256, 0, stream>>>(tmp, sa_ln_g + bo, sa_ln_b + bo, x, xbf);

    // ---- cross attention ----
    gemm_bf16_wmma<<<gO, 256, 0, stream>>>(xbf, wcaq + wo, ca_bq + bo, nullptr,
                                           nullptr, qbf, nullptr, M_, D_, D_, 0);
    gemm_bf16_wmma<<<gO, 256, 0, stream>>>(encbf, wcak + wo, ca_bk + bo, nullptr,
                                           nullptr, kbf, nullptr, M_, D_, D_, 0);
    gemm_bf16_wmma<<<gO, 256, 0, stream>>>(encbf, wcav + wo, ca_bv + bo, nullptr,
                                           nullptr, nullptr, vtbf, M_, D_, D_, 0);
    attn_wmma<<<gA, 128, 0, stream>>>(qbf, kbf, vtbf, enc_masks, ctxbf,
                                      LD_, LE_, 0, scale);
    gemm_bf16_wmma<<<gO, 256, 0, stream>>>(ctxbf, wcao + wo, ca_bo + bo, x,
                                           tmp, nullptr, nullptr, M_, D_, D_, 0);
    ln_kernel<<<M_, 256, 0, stream>>>(tmp, ca_ln_g + bo, ca_ln_b + bo, x, xbf);

    // ---- FFN ----
    gemm_bf16_wmma<<<gF, 256, 0, stream>>>(xbf, wff1 + w1o, ff_b1 + b1o, nullptr,
                                           nullptr, hbf, nullptr, M_, DFF_, D_, 1);
    gemm_bf16_wmma<<<gO, 256, 0, stream>>>(hbf, wff2 + w1o, ff_b2 + bo, x,
                                           tmp, nullptr, nullptr, M_, D_, DFF_, 0);
    ln_kernel<<<M_, 256, 0, stream>>>(tmp, ff_ln_g + bo, ff_ln_b + bo, x, xbf);
  }

  hipMemcpyAsync(d_out, x, (size_t)M_ * D_ * sizeof(float),
                 hipMemcpyDeviceToDevice, stream);
}
